// Attention_71373766525565
// MI455X (gfx1250) — matrix operand order repaired and checked by reading, not compiled, not run on hardware
//
#include <hip/hip_runtime.h>
#include <cstdint>

// ---------------------------------------------------------------------------
// Problem constants (reference: B=4, T=2048, C=1024, N_HEAD=16, hs=64)
// ---------------------------------------------------------------------------
#define BATCH 4
#define SEQ   2048
#define CH    1024
#define NHEAD 16
#define HS    64
#define MROWS (BATCH * SEQ)   // 8192

typedef unsigned short u16;
typedef unsigned int   u32;
typedef unsigned long long u64;

typedef __attribute__((ext_vector_type(16))) __bf16 v16bf;
typedef __attribute__((ext_vector_type(8)))  float  v8f;
typedef __attribute__((ext_vector_type(4)))  unsigned int v4u;
typedef __attribute__((ext_vector_type(8)))  int v8i;
typedef __attribute__((ext_vector_type(4)))  int v4i;

union Frag { u32 u[8]; v16bf v; };

__device__ __forceinline__ u16 f2bf(float f) {
    union { float f; u32 u; } c; c.f = f;
    u32 u = c.u;
    u32 r = (u + 0x7FFFu + ((u >> 16) & 1u)) >> 16;   // round-to-nearest-even
    return (u16)r;
}

__device__ __forceinline__ v8f vzero8() {
    v8f z = {0.f, 0.f, 0.f, 0.f, 0.f, 0.f, 0.f, 0.f};
    return z;
}

// ---------------------------------------------------------------------------
// Tensor Data Mover: async 2-D tile load (global -> LDS) with LDS padding.
// D# per CDNA5 ISA sec.8:
//   group0: count=1 | lds_addr | global_addr(57b) | type=2
//   group1: data_size=1(2B), pad_enable, pad_interval code, pad_amount code,
//           tensor_dim0/1, tile_dim0/1, tensor_dim0_stride
// Issue from one wave only; completion tracked with TENSORcnt.
// ---------------------------------------------------------------------------
__device__ __forceinline__ void tdm_load_2d(const void* gptr, u32 lds_off,
                                            u32 tensor_d0, u32 tensor_d1,
                                            u32 tile_d0, u32 tile_d1,
                                            u32 stride0,
                                            u32 pad_interval_code,
                                            u32 pad_amount_code) {
    u64 ga = (u64)(uintptr_t)gptr;
    v4u g0;
    g0.x = 1u;                                    // count=1, user descriptor
    g0.y = lds_off;                               // LDS byte address
    g0.z = (u32)ga;                               // global_addr[31:0]
    g0.w = ((u32)(ga >> 32) & 0x01FFFFFFu) | (2u << 30);  // addr[56:32] | type=2
    v8i g1;
    g1[0] = (int)((1u << 16)                      // data_size = 2 bytes
                | (1u << 20)                      // pad_enable
                | (pad_interval_code << 22)
                | (pad_amount_code << 25));
    g1[1] = (int)(tensor_d0 << 16);               // tensor_dim0[15:0]
    g1[2] = (int)((tensor_d0 >> 16) | (tensor_d1 << 16));
    g1[3] = (int)((tensor_d1 >> 16) | (tile_d0 << 16));
    g1[4] = (int)(tile_d1 & 0xFFFFu);             // tile_dim1 (tile_dim2 = 0)
    g1[5] = (int)stride0;                         // tensor_dim0_stride[31:0]
    g1[6] = 0;
    g1[7] = 0;
    v4i g2 = {0, 0, 0, 0};
    v4i g3 = {0, 0, 0, 0};
#if defined(__clang_major__) && (__clang_major__ >= 23)
    v8i g4 = {0, 0, 0, 0, 0, 0, 0, 0};
    __builtin_amdgcn_tensor_load_to_lds(g0, g1, g2, g3, g4, 0);
#else
    __builtin_amdgcn_tensor_load_to_lds(g0, g1, g2, g3, 0);
#endif
}

// ---------------------------------------------------------------------------
// f32 -> bf16 conversion (4 elements / thread)
// ---------------------------------------------------------------------------
__global__ __launch_bounds__(256) void cvt_f32_bf16(const float* __restrict__ src,
                                                    u16* __restrict__ dst, int n) {
    int i = (blockIdx.x * blockDim.x + threadIdx.x) * 4;
    if (i + 3 < n) {
        float4 f = *(const float4*)(src + i);
        u16 o[4] = { f2bf(f.x), f2bf(f.y), f2bf(f.z), f2bf(f.w) };
        *(uint2*)(dst + i) = *(const uint2*)o;
    }
}

// ---------------------------------------------------------------------------
// Y[m,n] = sum_k A[m,k] * W[n,k]   (y = A * W^T), bf16 in / f32 accumulate.
// Tile 128x128, BK=64, double-buffered LDS filled by the TDM; 8 waves (2x4),
// each wave 64x32 -> 8 accumulators, 16 WMMAs per K-step.
// Pipeline: TDM(i+1) issued right after the barrier publishing tile i.
// ---------------------------------------------------------------------------
template <int F32OUT>
__global__ __launch_bounds__(256) void gemm_bf16(const u16* __restrict__ A,
                                                 const u16* __restrict__ W,
                                                 void* __restrict__ Cout,
                                                 int M, int N, int K) {
    __shared__ u16 As[2][128 * 72];   // row stride 72 via TDM pad (64 + 8)
    __shared__ u16 Bs[2][128 * 72];

    const int tid  = threadIdx.x;
    const int wave = tid >> 5;
    const int lane = tid & 31;
    const int h    = lane >> 4;
    const int ln   = lane & 15;
    const int wm   = wave >> 2;    // 0..1
    const int wn   = wave & 3;     // 0..3
    const int m0   = blockIdx.y * 128;
    const int n0   = blockIdx.x * 128;

    const u32 aoff[2] = { (u32)(size_t)(void*)&As[0][0], (u32)(size_t)(void*)&As[1][0] };
    const u32 boff[2] = { (u32)(size_t)(void*)&Bs[0][0], (u32)(size_t)(void*)&Bs[1][0] };

    v8f acc[4][2];
#pragma unroll
    for (int i = 0; i < 4; ++i)
#pragma unroll
        for (int j = 0; j < 2; ++j) acc[i][j] = vzero8();

    const int nIter = K / 64;
    if (wave == 0) {
        tdm_load_2d(&A[(size_t)m0 * K], aoff[0], K, M, 64, 128, K, 4, 3);
        tdm_load_2d(&W[(size_t)n0 * K], boff[0], K, N, 64, 128, K, 4, 3);
    }

    for (int it = 0; it < nIter; ++it) {
        if (wave == 0) __builtin_amdgcn_s_wait_tensorcnt(0);
        __syncthreads();                              // tile `it` is visible
        if (wave == 0 && it + 1 < nIter) {            // prefetch tile it+1 via TDM
            int kb = (it + 1) * 64;
            tdm_load_2d(&A[(size_t)m0 * K + kb], aoff[(it + 1) & 1], K, M, 64, 128, K, 4, 3);
            tdm_load_2d(&W[(size_t)n0 * K + kb], boff[(it + 1) & 1], K, N, 64, 128, K, 4, 3);
        }

        const u16* Asb = &As[it & 1][0];
        const u16* Bsb = &Bs[it & 1][0];
#pragma unroll
        for (int kk = 0; kk < 2; ++kk) {
            Frag bfr[2];
#pragma unroll
            for (int jj = 0; jj < 2; ++jj) {
                int br = wn * 32 + jj * 16 + ln;      // N = lane within half
#pragma unroll
                for (int j = 0; j < 8; ++j) {
                    int k = kk * 32 + ((j < 4) ? 2 * j : 16 + 2 * (j - 4)) + 8 * h; // B: K-pairs, same order as A
                    bfr[jj].u[j] = *(const u32*)(&Bsb[br * 72 + k]);
                }
            }
#pragma unroll
            for (int i = 0; i < 4; ++i) {
                int ar = wm * 64 + i * 16 + ln;       // M = lane within half
                Frag af;
#pragma unroll
                for (int j = 0; j < 8; ++j) {
                    int k = kk * 32 + ((j < 4) ? 2 * j : 16 + 2 * (j - 4)) + 8 * h;
                    af.u[j] = *(const u32*)(&Asb[ar * 72 + k]);
                }
#pragma unroll
                for (int jj = 0; jj < 2; ++jj) {
                    acc[i][jj] = __builtin_amdgcn_wmma_f32_16x16x32_bf16(
                        false, af.v, false, bfr[jj].v, (short)0, acc[i][jj], false, false);
                }
            }
        }
        // No trailing barrier: next iteration's wait+barrier orders reuse; the
        // TDM issued above writes the buffer everyone stopped reading at the
        // previous barrier.
    }

    // C/D layout: VGPR r, half h -> row r + 8h; lane within half -> column.
#pragma unroll
    for (int i = 0; i < 4; ++i)
#pragma unroll
        for (int jj = 0; jj < 2; ++jj)
#pragma unroll
            for (int r = 0; r < 8; ++r) {
                int row = m0 + wm * 64 + i * 16 + r + 8 * h;
                int col = n0 + wn * 32 + jj * 16 + ln;
                if (F32OUT)
                    ((float*)Cout)[(size_t)row * N + col] = acc[i][jj][r];
                else
                    ((u16*)Cout)[(size_t)row * N + col] = f2bf(acc[i][jj][r]);
            }
}

// ---------------------------------------------------------------------------
// Fused flash attention (no mask in reference): per (b,h), O = softmax(QK^T/8)V
// Grid: (B*NHEAD, SEQ/128). Block 256 = 8 waves; wave owns 16 q-rows x 128 keys.
// K tile staged by TDM (padded to stride 72); V staged manually transposed.
// ---------------------------------------------------------------------------
__global__ __launch_bounds__(256) void flash_attn(const u16* __restrict__ Q,
                                                  const u16* __restrict__ Kb,
                                                  const u16* __restrict__ Vb,
                                                  u16* __restrict__ O) {
    __shared__ u16 Ks[128 * 72];        // [key][d]   stride 72 (TDM pad)
    __shared__ u16 Vts[64 * 136];       // [d][key]   stride 136 (V transposed)
    __shared__ u16 Ps[8][16 * 136];     // per-wave P [row][key]

    const int tid  = threadIdx.x;
    const int wave = tid >> 5;
    const int lane = tid & 31;
    const int h    = lane >> 4;
    const int ln   = lane & 15;
    const int b    = blockIdx.x >> 4;
    const int hh   = blockIdx.x & 15;
    const int qbase = blockIdx.y * 128 + wave * 16;

    const u32 ksOff = (u32)(size_t)(void*)&Ks[0];

    // Q fragments (16x64 -> two 16x32 A-frags), loaded once from global.
    Frag qf[2];
    {
        const u16* qp = Q + (size_t)(b * SEQ + qbase + ln) * CH + hh * HS;
#pragma unroll
        for (int kk = 0; kk < 2; ++kk)
#pragma unroll
            for (int j = 0; j < 8; ++j) {
                int k = kk * 32 + ((j < 4) ? 2 * j : 16 + 2 * (j - 4)) + 8 * h;
                qf[kk].u[j] = *(const u32*)(qp + k);
            }
    }

    float runMax[8], runSum[8];
    v8f oacc[4];
#pragma unroll
    for (int r = 0; r < 8; ++r) { runMax[r] = -1e30f; runSum[r] = 0.f; }
#pragma unroll
    for (int dt = 0; dt < 4; ++dt) oacc[dt] = vzero8();

    for (int kt = 0; kt < SEQ / 128; ++kt) {
        const int k0 = kt * 128;
        // --- stage K tile via TDM; V tile manually transposed ---
        if (wave == 0)
            tdm_load_2d(&Kb[(size_t)(b * SEQ + k0) * CH + hh * HS], ksOff,
                        CH, BATCH * SEQ, HS, 128, CH, 4, 3);
#pragma unroll
        for (int it = 0; it < 4; ++it) {
            int idx = (tid + it * 256) * 8;      // 0..8191
            int row = idx >> 6;                  // key
            int col = idx & 63;                  // d
            union { uint4 q; u16 s[8]; } t;
            t.q = *(const uint4*)(&Vb[(size_t)(b * SEQ + k0 + row) * CH + hh * HS + col]);
#pragma unroll
            for (int e = 0; e < 8; ++e) Vts[(col + e) * 136 + row] = t.s[e];
        }
        if (wave == 0) __builtin_amdgcn_s_wait_tensorcnt(0);
        __syncthreads();

        // --- S = Q K^T (8 N-tiles of 16 keys, contract over d=64) ---
        v8f s[8];
#pragma unroll
        for (int jt = 0; jt < 8; ++jt) s[jt] = vzero8();
#pragma unroll
        for (int jt = 0; jt < 8; ++jt) {
#pragma unroll
            for (int kk = 0; kk < 2; ++kk) {
                Frag bf;
#pragma unroll
                for (int j = 0; j < 8; ++j) {
                    int d = kk * 32 + ((j < 4) ? 2 * j : 16 + 2 * (j - 4)) + 8 * h;
                    bf.u[j] = *(const u32*)(&Ks[(jt * 16 + ln) * 72 + d]);
                }
                s[jt] = __builtin_amdgcn_wmma_f32_16x16x32_bf16(
                    false, qf[kk].v, false, bf.v, (short)0, s[jt], false, false);
            }
        }

        // --- online softmax (rows live in 16-lane halves; row = r + 8h) ---
        const float sc = 0.125f;   // 1/sqrt(64)
        float tmax[8];
#pragma unroll
        for (int r = 0; r < 8; ++r) tmax[r] = -1e30f;
#pragma unroll
        for (int jt = 0; jt < 8; ++jt)
#pragma unroll
            for (int r = 0; r < 8; ++r) {
                float v = s[jt][r] * sc;
                s[jt][r] = v;
                tmax[r] = fmaxf(tmax[r], v);
            }
#pragma unroll
        for (int m = 1; m <= 8; m <<= 1)
#pragma unroll
            for (int r = 0; r < 8; ++r)
                tmax[r] = fmaxf(tmax[r], __shfl_xor(tmax[r], m, 32));

        float alpha[8];
#pragma unroll
        for (int r = 0; r < 8; ++r) {
            float nm = fmaxf(runMax[r], tmax[r]);
            alpha[r] = __expf(runMax[r] - nm);
            runMax[r] = nm;
        }
        float rs[8];
#pragma unroll
        for (int r = 0; r < 8; ++r) rs[r] = 0.f;
#pragma unroll
        for (int jt = 0; jt < 8; ++jt)
#pragma unroll
            for (int r = 0; r < 8; ++r) {
                float p = __expf(s[jt][r] - runMax[r]);
                s[jt][r] = p;
                rs[r] += p;
            }
#pragma unroll
        for (int m = 1; m <= 8; m <<= 1)
#pragma unroll
            for (int r = 0; r < 8; ++r) rs[r] += __shfl_xor(rs[r], m, 32);
#pragma unroll
        for (int r = 0; r < 8; ++r) runSum[r] = runSum[r] * alpha[r] + rs[r];
#pragma unroll
        for (int dt = 0; dt < 4; ++dt)
#pragma unroll
            for (int r = 0; r < 8; ++r) oacc[dt][r] *= alpha[r];

        // --- P (C-layout) -> LDS -> A-layout fragments ---
#pragma unroll
        for (int jt = 0; jt < 8; ++jt)
#pragma unroll
            for (int r = 0; r < 8; ++r)
                Ps[wave][(r + 8 * h) * 136 + jt * 16 + ln] = f2bf(s[jt][r]);
        __syncthreads();

        // --- O += P V  (contract over 128 keys = 4 A-frags) ---
#pragma unroll
        for (int jj = 0; jj < 4; ++jj) {
            Frag af;
#pragma unroll
            for (int j = 0; j < 8; ++j) {
                int k = jj * 32 + ((j < 4) ? 2 * j : 16 + 2 * (j - 4)) + 8 * h;
                af.u[j] = *(const u32*)(&Ps[wave][ln * 136 + k]);
            }
#pragma unroll
            for (int dt = 0; dt < 4; ++dt) {
                Frag bf;
#pragma unroll
                for (int j = 0; j < 8; ++j) {
                    int key = jj * 32 + ((j < 4) ? 2 * j : 16 + 2 * (j - 4)) + 8 * h;
                    bf.u[j] = *(const u32*)(&Vts[(dt * 16 + ln) * 136 + key]);
                }
                oacc[dt] = __builtin_amdgcn_wmma_f32_16x16x32_bf16(
                    false, af.v, false, bf.v, (short)0, oacc[dt], false, false);
            }
        }
        __syncthreads();
    }

    // --- normalize and store bf16 ---
#pragma unroll
    for (int dt = 0; dt < 4; ++dt)
#pragma unroll
        for (int r = 0; r < 8; ++r) {
            int row = qbase + r + 8 * h;
            int d   = dt * 16 + ln;
            float v = oacc[dt][r] / runSum[r];
            O[(size_t)(b * SEQ + row) * CH + hh * HS + d] = f2bf(v);
        }
}

// ---------------------------------------------------------------------------
// Launch: cvt(x, W*) -> Q/K/V GEMMs -> flash attention -> output GEMM (f32)
// ---------------------------------------------------------------------------
extern "C" void kernel_launch(void* const* d_in, const int* in_sizes, int n_in,
                              void* d_out, int out_size, void* d_ws, size_t ws_size,
                              hipStream_t stream) {
    const float* x  = (const float*)d_in[0];
    const float* Wk = (const float*)d_in[1];
    const float* Wq = (const float*)d_in[2];
    const float* Wv = (const float*)d_in[3];
    const float* Wo = (const float*)d_in[4];

    char* ws = (char*)d_ws;
    const size_t XB  = (size_t)MROWS * CH * 2;   // 16 MB bf16
    const size_t WB  = (size_t)CH * CH * 2;      // 2 MB bf16
    u16* xb  = (u16*)(ws);
    u16* Wkb = (u16*)(ws + XB);
    u16* Wqb = (u16*)(ws + XB + 1 * WB);
    u16* Wvb = (u16*)(ws + XB + 2 * WB);
    u16* Wob = (u16*)(ws + XB + 3 * WB);
    u16* Qb  = (u16*)(ws + XB + 4 * WB);
    u16* Kb  = (u16*)(ws + XB + 4 * WB + 1 * XB);
    u16* Vb  = (u16*)(ws + XB + 4 * WB + 2 * XB);
    u16* Ob  = (u16*)(ws + XB + 4 * WB + 3 * XB);

    const int nX = MROWS * CH;   // 8388608
    const int nW = CH * CH;      // 1048576
    cvt_f32_bf16<<<nX / (256 * 4), 256, 0, stream>>>(x, xb, nX);
    cvt_f32_bf16<<<nW / (256 * 4), 256, 0, stream>>>(Wk, Wkb, nW);
    cvt_f32_bf16<<<nW / (256 * 4), 256, 0, stream>>>(Wq, Wqb, nW);
    cvt_f32_bf16<<<nW / (256 * 4), 256, 0, stream>>>(Wv, Wvb, nW);
    cvt_f32_bf16<<<nW / (256 * 4), 256, 0, stream>>>(Wo, Wob, nW);

    dim3 gGemm(CH / 128, MROWS / 128);   // (8, 64)
    gemm_bf16<0><<<gGemm, 256, 0, stream>>>(xb, Wqb, Qb, MROWS, CH, CH);
    gemm_bf16<0><<<gGemm, 256, 0, stream>>>(xb, Wkb, Kb, MROWS, CH, CH);
    gemm_bf16<0><<<gGemm, 256, 0, stream>>>(xb, Wvb, Vb, MROWS, CH, CH);

    dim3 gAttn(BATCH * NHEAD, SEQ / 128);   // (64, 16)
    flash_attn<<<gAttn, 256, 0, stream>>>(Qb, Kb, Vb, Ob);

    gemm_bf16<1><<<gGemm, 256, 0, stream>>>(Ob, Wob, d_out, MROWS, CH, CH);
}
